// RGATModel_86371792322702
// MI455X (gfx1250) — matrix operand order repaired and checked by reading, not compiled, not run on hardware
//
#include <hip/hip_runtime.h>
#include <hip/hip_bf16.h>

#define N_NODES 50000
#define N_EDGES 320000
#define N_REL   3
#define DIM_IN  768
#define DIM_HID 256
#define DIM_OUT 64
#define NEG_SLOPE 0.2f

typedef __attribute__((ext_vector_type(16))) __bf16 v16bf;
typedef __attribute__((ext_vector_type(8)))  __bf16 v8bf;
typedef __attribute__((ext_vector_type(4)))  __bf16 v4bf;
typedef __attribute__((ext_vector_type(8)))  float  v8f;

__device__ __forceinline__ __bf16 f2bf(float f) {
    unsigned u = __float_as_uint(f);
    u += 0x7FFFu + ((u >> 16) & 1u);          // round-to-nearest-even
    unsigned short s = (unsigned short)(u >> 16);
    return __builtin_bit_cast(__bf16, s);
}

// ---------------------------------------------------------------------------
// C = act(A[M,K] @ B[K,Nc] + bias). A is bf16 row-major, BT is bf16 [Nc,K]
// (pre-transposed weights, L2-resident). f32 WMMA accumulate.
// Block = 256 threads = 8 waves; wave w owns a 16x64 output strip.
// Barrier-free. Per 32-K step: 2 A b128 loads + 8 B b128 loads issued as a
// batch, then 4 WMMAs (so waits drain partially, overlapping L2 latency).
// K compile-time (multiple of 32), Nc % 64 == 0, M % 16 == 0.
// ACT: 1 = relu. WMODE bit0: write f32 C; bit1: write bf16 Cbf.
// ---------------------------------------------------------------------------
template<int K, int ACT, int WMODE>
__global__ __launch_bounds__(256)
void gemm_bf16_wmma(const __bf16* __restrict__ A, const __bf16* __restrict__ BT,
                    const float* __restrict__ bias, float* __restrict__ C,
                    __bf16* __restrict__ Cbf, int M, int Nc) {
    const int wave    = threadIdx.x >> 5;
    const int lane    = threadIdx.x & 31;
    const int colBase = blockIdx.x * 64;
    const int tileM   = blockIdx.y * 8 + wave;
    if (tileM >= (M >> 4)) return;                  // wave-uniform, no barriers

    const int m  = lane & 15;       // row within A tile / col within subtile
    const int hi = lane >> 4;       // lane half
    const int kA = hi << 3;         // A frag K base within 32-step: 0 or 8
    const int kB = hi << 3;         // B frag K base within 32-step: 0 or 8

    const __bf16* Arow  = A  + (size_t)(tileM * 16 + m) * K;
    const __bf16* Brow0 = BT + (size_t)(colBase + m) * K;

    v8f acc[4];
    #pragma unroll
    for (int nt = 0; nt < 4; ++nt)
        #pragma unroll
        for (int j = 0; j < 8; ++j) acc[nt][j] = 0.0f;

    #pragma unroll 4
    for (int k0 = 0; k0 < K; k0 += 32) {
        // ---- issue all loads for this k-step first ----
        v8bf a0 = *(const v8bf*)(Arow + k0 + kA);
        v8bf a1 = *(const v8bf*)(Arow + k0 + 16 + kA);
        v8bf bb[8];
        #pragma unroll
        for (int nt = 0; nt < 4; ++nt) {
            const __bf16* bp = Brow0 + (size_t)nt * 16 * K + k0 + kB;
            bb[2 * nt]     = *(const v8bf*)(bp);
            bb[2 * nt + 1] = *(const v8bf*)(bp + 16);
        }
        if (k0 + 128 < K) __builtin_prefetch(Arow + k0 + 128, 0, 1);

        // ---- then consume: 4 WMMAs ----
        v16bf afrag = __builtin_shufflevector(a0, a1,
            0, 1, 2, 3, 4, 5, 6, 7, 8, 9, 10, 11, 12, 13, 14, 15);
        #pragma unroll
        for (int nt = 0; nt < 4; ++nt) {
            v16bf bfrag = __builtin_shufflevector(bb[2 * nt], bb[2 * nt + 1],
                0, 1, 2, 3, 4, 5, 6, 7, 8, 9, 10, 11, 12, 13, 14, 15);
            acc[nt] = __builtin_amdgcn_wmma_f32_16x16x32_bf16(
                false, afrag, false, bfrag, (short)0, acc[nt], false, false);
        }
    }

    #pragma unroll
    for (int nt = 0; nt < 4; ++nt) {
        const int col = colBase + nt * 16 + m;
        const float bv = bias ? bias[col] : 0.0f;
        #pragma unroll
        for (int v = 0; v < 8; ++v) {
            int row = tileM * 16 + v + (hi << 3);   // C/D lane layout
            float x = acc[nt][v] + bv;
            if (ACT == 1) x = x > 0.0f ? x : 0.0f;
            if (WMODE & 1) C[(size_t)row * Nc + col] = x;
            if (WMODE & 2) Cbf[(size_t)row * Nc + col] = f2bf(x);
        }
    }
}

// dst[i] = bf16(src[i]), vectorized by 4
__global__ __launch_bounds__(256)
void f32_to_bf16(const float* __restrict__ s, __bf16* __restrict__ d, int n4) {
    int i = blockIdx.x * blockDim.x + threadIdx.x;
    if (i >= n4) return;
    float4 v = ((const float4*)s)[i];
    v4bf o;
    o[0] = f2bf(v.x); o[1] = f2bf(v.y); o[2] = f2bf(v.z); o[3] = f2bf(v.w);
    ((v4bf*)d)[i] = o;
}

// BT[c, k] = bf16(B[k, c])   (B is [K, Nc] f32 row-major)
__global__ __launch_bounds__(256)
void transpose_to_bf16(const float* __restrict__ B, __bf16* __restrict__ BT,
                       int K, int Nc) {
    int gid = blockIdx.x * blockDim.x + threadIdx.x;
    if (gid >= K * Nc) return;
    int k = gid / Nc, c = gid - k * Nc;
    BT[(size_t)c * K + k] = f2bf(B[gid]);
}

// ---------------------------------------------------------------------------
// Per-row dots with q and k: oq[row] = xw[row,:]·q ; ok[row] = xw[row,:]·k
// ---------------------------------------------------------------------------
__global__ __launch_bounds__(256)
void rowdot_qk(const float* __restrict__ xw, const float* __restrict__ q,
               const float* __restrict__ k, float* __restrict__ oq,
               float* __restrict__ ok, int rows) {
    int wave = (blockIdx.x * blockDim.x + threadIdx.x) >> 5;
    int lane = threadIdx.x & 31;
    if (wave >= rows) return;
    const float* row = xw + (size_t)wave * DIM_HID;
    int base = lane * 8;
    float sq = 0.0f, sk = 0.0f;
    #pragma unroll
    for (int i = 0; i < 8; i += 4) {
        float4 v  = *(const float4*)(row + base + i);
        float4 qv = *(const float4*)(q + base + i);
        float4 kv = *(const float4*)(k + base + i);
        sq += v.x * qv.x + v.y * qv.y + v.z * qv.z + v.w * qv.w;
        sk += v.x * kv.x + v.y * kv.y + v.z * kv.z + v.w * kv.w;
    }
    #pragma unroll
    for (int off = 16; off > 0; off >>= 1) {
        sq += __shfl_down(sq, off, 32);
        sk += __shfl_down(sk, off, 32);
    }
    if (lane == 0) { oq[wave] = sq; ok[wave] = sk; }
}

// out[n,:] = bias ; mx[n] = -inf ; den[n] = 0
__global__ __launch_bounds__(256)
void seg_init(float* __restrict__ out, float* __restrict__ mx,
              float* __restrict__ den, const float* __restrict__ bias) {
    int node = blockIdx.x;
    int c = threadIdx.x;
    out[(size_t)node * DIM_HID + c] = bias[c];
    if (c == 0) { mx[node] = -3.0e38f; den[node] = 0.0f; }
}

__device__ __forceinline__ void atomicMaxF(float* addr, float v) {
    unsigned* ua = (unsigned*)addr;
    unsigned old = *ua;
    while (__uint_as_float(old) < v) {
        unsigned assumed = old;
        old = atomicCAS(ua, assumed, __float_as_uint(v));
        if (old == assumed) break;
    }
}

// alpha_e = leaky_relu(xwq[et,dst] + xwk[et,src]); mx[dst] = max(mx[dst], alpha_e)
__global__ __launch_bounds__(256)
void edge_alpha(const int* __restrict__ src, const int* __restrict__ dst,
                const int* __restrict__ et, const float* __restrict__ xwq,
                const float* __restrict__ xwk, float* __restrict__ alpha,
                float* __restrict__ mx) {
    int e = blockIdx.x * blockDim.x + threadIdx.x;
    if (e >= N_EDGES) return;
    int s = src[e], d = dst[e], t = et[e];
    float a = xwq[(size_t)t * N_NODES + d] + xwk[(size_t)t * N_NODES + s];
    a = a > 0.0f ? a : NEG_SLOPE * a;
    alpha[e] = a;
    atomicMaxF(&mx[d], a);
}

// ea_e = exp(alpha_e - mx[dst]); den[dst] += ea_e
__global__ __launch_bounds__(256)
void edge_exp(const int* __restrict__ dst, const float* __restrict__ alpha,
              const float* __restrict__ mx, float* __restrict__ ea,
              float* __restrict__ den) {
    int e = blockIdx.x * blockDim.x + threadIdx.x;
    if (e >= N_EDGES) return;
    int d = dst[e];
    float v = __expf(alpha[e] - mx[d]);
    ea[e] = v;
    atomicAdd(&den[d], v);
}

// out[dst,:] += (ea_e / (den[dst]+1e-16)) * xw[et, src, :]
__global__ __launch_bounds__(256)
void edge_aggregate(const int* __restrict__ src, const int* __restrict__ dst,
                    const int* __restrict__ et, const float* __restrict__ xw,
                    const float* __restrict__ ea, const float* __restrict__ den,
                    float* __restrict__ out) {
    int gid = blockIdx.x * blockDim.x + threadIdx.x;
    int e = gid >> 6;
    int c = (gid & 63) << 2;
    if (e >= N_EDGES) return;
    int s = src[e], d = dst[e], t = et[e];
    float a = ea[e] / (den[d] + 1e-16f);
    float4 v = *(const float4*)(xw + ((size_t)t * N_NODES + s) * DIM_HID + c);
    float* o = out + (size_t)d * DIM_HID + c;
    atomicAdd(o + 0, a * v.x);
    atomicAdd(o + 1, a * v.y);
    atomicAdd(o + 2, a * v.z);
    atomicAdd(o + 3, a * v.w);
}

// logits = h[N,64] @ w_cls[64,2] + b_cls
__global__ __launch_bounds__(256)
void cls_head(const float* __restrict__ h, const float* __restrict__ w,
              const float* __restrict__ b, float* __restrict__ out) {
    int i = blockIdx.x * blockDim.x + threadIdx.x;
    if (i >= N_NODES) return;
    float s0 = b[0], s1 = b[1];
    const float* row = h + (size_t)i * DIM_OUT;
    #pragma unroll 4
    for (int c = 0; c < DIM_OUT; c += 4) {
        float4 v = *(const float4*)(row + c);
        s0 += v.x * w[(c + 0) * 2] + v.y * w[(c + 1) * 2] + v.z * w[(c + 2) * 2] + v.w * w[(c + 3) * 2];
        s1 += v.x * w[(c + 0) * 2 + 1] + v.y * w[(c + 1) * 2 + 1] + v.z * w[(c + 2) * 2 + 1] + v.w * w[(c + 3) * 2 + 1];
    }
    out[(size_t)i * 2 + 0] = s0;
    out[(size_t)i * 2 + 1] = s1;
}

extern "C" void kernel_launch(void* const* d_in, const int* in_sizes, int n_in,
                              void* d_out, int out_size, void* d_ws, size_t ws_size,
                              hipStream_t stream) {
    const float* x     = (const float*)d_in[0];
    const int*   ei    = (const int*)d_in[1];
    const int*   etyp  = (const int*)d_in[2];
    const float* w_in  = (const float*)d_in[3];
    const float* b_in  = (const float*)d_in[4];
    const float* c1w   = (const float*)d_in[5];
    const float* c1q   = (const float*)d_in[6];
    const float* c1k   = (const float*)d_in[7];
    const float* c1b   = (const float*)d_in[8];
    const float* c2w   = (const float*)d_in[9];
    const float* c2q   = (const float*)d_in[10];
    const float* c2k   = (const float*)d_in[11];
    const float* c2b   = (const float*)d_in[12];
    const float* w_out = (const float*)d_in[13];
    const float* b_out = (const float*)d_in[14];
    const float* w_cls = (const float*)d_in[15];
    const float* b_cls = (const float*)d_in[16];

    const int* src = ei;             // edge_index[0,:]
    const int* dst = ei + N_EDGES;   // edge_index[1,:]

    // ---- workspace carving: f32 region then bf16 region ----
    float* ws    = (float*)d_ws;
    float* h1f   = ws;                                   // N*HID   (layer1 out)
    float* h2f   = h1f + (size_t)N_NODES * DIM_HID;      // N*HID   (layer2 out)
    float* xw    = h2f + (size_t)N_NODES * DIM_HID;      // R*N*HID
    float* xwq   = xw  + (size_t)N_REL * N_NODES * DIM_HID;
    float* xwk   = xwq + (size_t)N_REL * N_NODES;
    float* alpha = xwk + (size_t)N_REL * N_NODES;        // E
    float* ea    = alpha + N_EDGES;                      // E
    float* mx    = ea + N_EDGES;                         // N
    float* den   = mx + N_NODES;                         // N
    float* h3    = den + N_NODES;                        // N*OUT
    float* f32end = h3 + (size_t)N_NODES * DIM_OUT;

    __bf16* bws   = (__bf16*)f32end;
    __bf16* x_bf  = bws;                                  // N*IN
    __bf16* hbA   = x_bf + (size_t)N_NODES * DIM_IN;      // N*HID (in-proj / layer2 out)
    __bf16* hbB   = hbA  + (size_t)N_NODES * DIM_HID;     // N*HID (layer1 out)
    __bf16* wT_in = hbB  + (size_t)N_NODES * DIM_HID;     // HID x IN
    __bf16* wT1   = wT_in + (size_t)DIM_HID * DIM_IN;     // R x (HID x HID)
    __bf16* wT2   = wT1   + (size_t)N_REL * DIM_HID * DIM_HID;
    __bf16* wT_out= wT2   + (size_t)N_REL * DIM_HID * DIM_HID; // OUT x HID

    const int Mtiles = N_NODES / 16;                     // 3125
    const dim3 gHid(DIM_HID / 64, (Mtiles + 7) / 8);     // (4, 391)
    const dim3 gOut(DIM_OUT / 64, (Mtiles + 7) / 8);     // (1, 391)
    const int edgeBlocks = (N_EDGES + 255) / 256;
    const int aggBlocks  = (N_EDGES * 64) / 256;
    const int rows = N_REL * N_NODES;
    const int rowdotBlocks = (rows + 7) / 8;
    const int hConv4 = (N_NODES * DIM_HID) / 4;

    // ---- one-time conversions / weight transposes ----
    f32_to_bf16<<<(N_NODES * DIM_IN / 4 + 255) / 256, 256, 0, stream>>>(
        x, x_bf, N_NODES * DIM_IN / 4);
    transpose_to_bf16<<<(DIM_IN * DIM_HID + 255) / 256, 256, 0, stream>>>(
        w_in, wT_in, DIM_IN, DIM_HID);
    for (int r = 0; r < N_REL; ++r) {
        transpose_to_bf16<<<(DIM_HID * DIM_HID + 255) / 256, 256, 0, stream>>>(
            c1w + (size_t)r * DIM_HID * DIM_HID, wT1 + (size_t)r * DIM_HID * DIM_HID,
            DIM_HID, DIM_HID);
        transpose_to_bf16<<<(DIM_HID * DIM_HID + 255) / 256, 256, 0, stream>>>(
            c2w + (size_t)r * DIM_HID * DIM_HID, wT2 + (size_t)r * DIM_HID * DIM_HID,
            DIM_HID, DIM_HID);
    }
    transpose_to_bf16<<<(DIM_HID * DIM_OUT + 255) / 256, 256, 0, stream>>>(
        w_out, wT_out, DIM_HID, DIM_OUT);

    // ---- input projection: hbA = bf16(relu(x @ w_in + b_in)) ----
    gemm_bf16_wmma<DIM_IN, 1, 2><<<gHid, 256, 0, stream>>>(
        x_bf, wT_in, b_in, nullptr, hbA, N_NODES, DIM_HID);

    // ---- one RGAT layer ----
    auto rgat = [&](const __bf16* hin_bf, const __bf16* wT, const float* q,
                    const float* k, const float* b, float* houtF, __bf16* hout_bf) {
        for (int r = 0; r < N_REL; ++r)
            gemm_bf16_wmma<DIM_HID, 0, 1><<<gHid, 256, 0, stream>>>(
                hin_bf, wT + (size_t)r * DIM_HID * DIM_HID, nullptr,
                xw + (size_t)r * N_NODES * DIM_HID, nullptr, N_NODES, DIM_HID);
        rowdot_qk<<<rowdotBlocks, 256, 0, stream>>>(xw, q, k, xwq, xwk, rows);
        seg_init<<<N_NODES, DIM_HID, 0, stream>>>(houtF, mx, den, b);
        edge_alpha<<<edgeBlocks, 256, 0, stream>>>(src, dst, etyp, xwq, xwk, alpha, mx);
        edge_exp<<<edgeBlocks, 256, 0, stream>>>(dst, alpha, mx, ea, den);
        edge_aggregate<<<aggBlocks, 256, 0, stream>>>(src, dst, etyp, xw, ea, den, houtF);
        f32_to_bf16<<<(hConv4 + 255) / 256, 256, 0, stream>>>(houtF, hout_bf, hConv4);
    };

    rgat(hbA, wT1, c1q, c1k, c1b, h1f, hbB);
    rgat(hbB, wT2, c2q, c2k, c2b, h2f, hbA);   // reuse hbA for layer2 output

    // ---- output projection: h3 = relu(h2 @ w_out + b_out) ----
    gemm_bf16_wmma<DIM_HID, 1, 1><<<gOut, 256, 0, stream>>>(
        hbA, wT_out, b_out, h3, nullptr, N_NODES, DIM_OUT);

    // ---- classifier ----
    cls_head<<<(N_NODES + 255) / 256, 256, 0, stream>>>(h3, w_cls, b_cls,
                                                        (float*)d_out);
}
